// MyRobustGINModel_12180527252134
// MI455X (gfx1250) — matrix operand order repaired and checked by reading, not compiled, not run on hardware
//
#include <hip/hip_runtime.h>

#define NNODES 50000
#define NEDGES 800000
#define NGRAPH 512

typedef __attribute__((ext_vector_type(16))) __bf16 v16bf;
typedef __attribute__((ext_vector_type(8)))  float  v8f;

union ABu { v16bf v; unsigned u[8]; };

// round-to-nearest-even f32 -> bf16, packed pair (low = first element)
__device__ inline unsigned pack2(float a, float b) {
    unsigned ua = __builtin_bit_cast(unsigned, a);
    unsigned ub = __builtin_bit_cast(unsigned, b);
    ua = ua + 0x7FFFu + ((ua >> 16) & 1u);
    ub = ub + 0x7FFFu + ((ub >> 16) & 1u);
    return (ua >> 16) | (ub & 0xFFFF0000u);
}

// ---------------- fill ----------------
__global__ void k_fill0(float* __restrict__ p, int n) {
    int t = blockIdx.x * blockDim.x + threadIdx.x;
    if (t < n) p[t] = 0.0f;
}

// ---------------- weight pre-pack (f32 -> packed bf16 pairs, [col][k/2]) ----------------
__global__ void k_prep(const float* __restrict__ W1, const float* __restrict__ W2,
                       unsigned* __restrict__ w1p, unsigned* __restrict__ w2p) {
    int t = blockIdx.x * blockDim.x + threadIdx.x;
    if (t < 4 * 128 * 32) {                       // W1: [4][64][128] -> [4][c=128][k2=32]
        int i = t / (128 * 32); int r = t % (128 * 32);
        int c = r >> 5; int k2 = r & 31;
        const float* W = W1 + i * 64 * 128;
        w1p[t] = pack2(W[(2 * k2) * 128 + c], W[(2 * k2 + 1) * 128 + c]);
    } else if (t < 4 * 128 * 32 + 4 * 64 * 64) {  // W2: [4][128][64] -> [4][c=64][k2=64]
        int t2 = t - 4 * 128 * 32;
        int i = t2 / (64 * 64); int r = t2 % (64 * 64);
        int c = r >> 6; int k2 = r & 63;
        const float* W = W2 + i * 128 * 64;
        w2p[t2] = pack2(W[(2 * k2) * 64 + c], W[(2 * k2 + 1) * 64 + c]);
    }
}

// ---------------- embedding gather: h[n] = emb[x[n]] ----------------
__global__ void k_embed(const int* __restrict__ x, const float* __restrict__ emb,
                        float* __restrict__ h) {
    int t = blockIdx.x * blockDim.x + threadIdx.x;   // N*16 threads, float4 each
    if (t >= NNODES * 16) return;
    int n = t >> 4, c = (t & 15) * 4;
    *(float4*)(h + (size_t)n * 64 + c) = *(const float4*)(emb + (size_t)x[n] * 64 + c);
}

// ---------------- edge scatter-add: agg[dst] += h[src] ----------------
__global__ void k_edge(const int* __restrict__ ei, const float* __restrict__ h,
                       float* __restrict__ agg) {
    int t = blockIdx.x * blockDim.x + threadIdx.x;   // E*16 threads
    int e = t >> 4, c = (t & 15) * 4;
    if (e >= NEDGES) return;
    int src = ei[e];
    int dst = ei[NEDGES + e];
    float4 v = *(const float4*)(h + (size_t)src * 64 + c);
    float* a = agg + (size_t)dst * 64 + c;
    atomicAdd(a + 0, v.x); atomicAdd(a + 1, v.y);
    atomicAdd(a + 2, v.z); atomicAdd(a + 3, v.w);
}

// ---------------- GEMM1: y = (h+agg) @ W1 + b1, accumulate BN stats ----------------
__global__ __launch_bounds__(256) void k_gemm1(
    const float* __restrict__ h, const float* __restrict__ agg,
    const unsigned* __restrict__ w1pg, const float* __restrict__ b1,
    float* __restrict__ y, float* __restrict__ ssum, float* __restrict__ ssq)
{
    __shared__ unsigned zp[128][33];   // packed bf16 z tile, rows x k-pairs
    __shared__ unsigned wp[128][33];   // packed bf16 W1, cols x k-pairs
    const int tid = threadIdx.x;
    const int tile0 = blockIdx.x * 128;

    for (int j = 0; j < 16; ++j) {
        int idx = j * 256 + tid;              // 0..4095
        int row = idx >> 5, k2 = idx & 31;
        int gr = tile0 + row;
        float a0 = 0.f, a1 = 0.f;
        if (gr < NNODES) {
            float2 hv = *(const float2*)(h   + (size_t)gr * 64 + 2 * k2);
            float2 av = *(const float2*)(agg + (size_t)gr * 64 + 2 * k2);
            a0 = hv.x + av.x; a1 = hv.y + av.y;
        }
        zp[row][k2] = pack2(a0, a1);
        wp[row][k2] = w1pg[idx];
    }
    __syncthreads();

    const int wave = tid >> 5, lane = tid & 31;
    const int cl = lane & 15, hi = lane >> 4;
    const int rT = wave * 16;

    v8f acc[8] = {};
    #pragma unroll
    for (int ks = 0; ks < 2; ++ks) {
        ABu a;
        #pragma unroll
        for (int v = 0; v < 8; ++v) {
            int k2 = ks * 16 + v + (hi ? 8 : 0);
            a.u[v] = zp[rT + cl][k2];
        }
        #pragma unroll
        for (int nt = 0; nt < 8; ++nt) {
            ABu b;
            #pragma unroll
            for (int v = 0; v < 8; ++v) {
                int k2 = ks * 16 + v + (hi ? 8 : 0);
                b.u[v] = wp[nt * 16 + cl][k2];
            }
            acc[nt] = __builtin_amdgcn_wmma_f32_16x16x32_bf16(
                false, a.v, false, b.v, (short)0, acc[nt], false, false);
        }
    }

    #pragma unroll
    for (int nt = 0; nt < 8; ++nt) {
        int col = nt * 16 + cl;
        float bias = b1[col];
        float s = 0.f, q = 0.f;
        #pragma unroll
        for (int v = 0; v < 8; ++v) {
            int gr = tile0 + rT + v + hi * 8;
            float val = acc[nt][v] + bias;
            if (gr < NNODES) {
                y[(size_t)gr * 128 + col] = val;
                s += val; q += val * val;
            }
        }
        atomicAdd(&ssum[col], s);
        atomicAdd(&ssq[col], q);
    }
}

// ---------------- BN fold: scale = g*rsqrt(var+eps), shift = b - mu*scale ----------------
__global__ void k_bnfold(const float* __restrict__ ssum, const float* __restrict__ ssq,
                         const float* __restrict__ gamma, const float* __restrict__ beta,
                         float* __restrict__ scale, float* __restrict__ shift,
                         int C, float invN) {
    int c = threadIdx.x;
    if (c >= C) return;
    float mu  = ssum[c] * invN;
    float var = ssq[c] * invN - mu * mu;
    float sc  = gamma[c] * rsqrtf(var + 1e-5f);
    scale[c] = sc;
    shift[c] = beta[c] - mu * sc;
}

// ---------------- GEMM2: h' = relu( relu(BN1(y)) @ W2 + b2 ), accumulate BN2 stats ----------------
__global__ __launch_bounds__(256) void k_gemm2(
    const float* __restrict__ y,
    const float* __restrict__ scale1, const float* __restrict__ shift1,
    const unsigned* __restrict__ w2pg, const float* __restrict__ b2,
    float* __restrict__ hout, float* __restrict__ ssum, float* __restrict__ ssq,
    int do_stats)
{
    __shared__ unsigned yp[128][65];   // packed bf16 activations, rows x 64 k-pairs
    __shared__ unsigned wp[64][65];    // packed bf16 W2, 64 cols x 64 k-pairs
    __shared__ float s1[128], t1[128];
    const int tid = threadIdx.x;
    const int tile0 = blockIdx.x * 128;

    if (tid < 128) { s1[tid] = scale1[tid]; t1[tid] = shift1[tid]; }
    __syncthreads();

    for (int j = 0; j < 32; ++j) {
        int idx = j * 256 + tid;              // 0..8191
        int row = idx >> 6, k2 = idx & 63;
        int gr = tile0 + row;
        float a0 = 0.f, a1 = 0.f;
        if (gr < NNODES) {
            float2 v = *(const float2*)(y + (size_t)gr * 128 + 2 * k2);
            a0 = fmaxf(v.x * s1[2 * k2]     + t1[2 * k2],     0.f);
            a1 = fmaxf(v.y * s1[2 * k2 + 1] + t1[2 * k2 + 1], 0.f);
        }
        yp[row][k2] = pack2(a0, a1);
    }
    for (int j = 0; j < 16; ++j) {
        int idx = j * 256 + tid;              // 0..4095
        wp[idx >> 6][idx & 63] = w2pg[idx];
    }
    __syncthreads();

    const int wave = tid >> 5, lane = tid & 31;
    const int cl = lane & 15, hi = lane >> 4;
    const int rT = wave * 16;

    v8f acc[4] = {};
    #pragma unroll
    for (int ks = 0; ks < 4; ++ks) {
        ABu a;
        #pragma unroll
        for (int v = 0; v < 8; ++v) {
            int k2 = ks * 16 + v + (hi ? 8 : 0);
            a.u[v] = yp[rT + cl][k2];
        }
        #pragma unroll
        for (int nt = 0; nt < 4; ++nt) {
            ABu b;
            #pragma unroll
            for (int v = 0; v < 8; ++v) {
                int k2 = ks * 16 + v + (hi ? 8 : 0);
                b.u[v] = wp[nt * 16 + cl][k2];
            }
            acc[nt] = __builtin_amdgcn_wmma_f32_16x16x32_bf16(
                false, a.v, false, b.v, (short)0, acc[nt], false, false);
        }
    }

    #pragma unroll
    for (int nt = 0; nt < 4; ++nt) {
        int col = nt * 16 + cl;
        float bias = b2[col];
        float s = 0.f, q = 0.f;
        #pragma unroll
        for (int v = 0; v < 8; ++v) {
            int gr = tile0 + rT + v + hi * 8;
            float val = fmaxf(acc[nt][v] + bias, 0.f);
            if (gr < NNODES) {
                hout[(size_t)gr * 64 + col] = val;
                s += val; q += val * val;
            }
        }
        if (do_stats) {
            atomicAdd(&ssum[col], s);
            atomicAdd(&ssq[col], q);
        }
    }
}

// ---------------- elementwise BN apply on h [N,64] ----------------
__global__ void k_bnapply(float* __restrict__ h, const float* __restrict__ sc,
                          const float* __restrict__ sh) {
    int t = blockIdx.x * blockDim.x + threadIdx.x;
    if (t >= NNODES * 16) return;
    int n = t >> 4, c = (t & 15) * 4;
    float4 v = *(float4*)(h + (size_t)n * 64 + c);
    v.x = v.x * sc[c + 0] + sh[c + 0];
    v.y = v.y * sc[c + 1] + sh[c + 1];
    v.z = v.z * sc[c + 2] + sh[c + 2];
    v.w = v.w * sc[c + 3] + sh[c + 3];
    *(float4*)(h + (size_t)n * 64 + c) = v;
}

// ---------------- global_add_pool ----------------
__global__ void k_pool(const float* __restrict__ h, const int* __restrict__ batch,
                       float* __restrict__ gp) {
    int t = blockIdx.x * blockDim.x + threadIdx.x;
    if (t >= NNODES * 16) return;
    int n = t >> 4, c = (t & 15) * 4;
    int g = batch[n];
    float4 v = *(const float4*)(h + (size_t)n * 64 + c);
    float* d = gp + (size_t)g * 64 + c;
    atomicAdd(d + 0, v.x); atomicAdd(d + 1, v.y);
    atomicAdd(d + 2, v.z); atomicAdd(d + 3, v.w);
}

// ---------------- head MLP: out = relu(g@Wh1+bh1)@Wh2 + bh2 ----------------
__global__ __launch_bounds__(256) void k_head(
    const float* __restrict__ gp, const float* __restrict__ Wh1, const float* __restrict__ bh1,
    const float* __restrict__ Wh2, const float* __restrict__ bh2, float* __restrict__ out)
{
    __shared__ float gs[8][64];
    __shared__ float hs[8][32];
    const int tid = threadIdx.x;
    const int gbase = blockIdx.x * 8;
    for (int j = tid; j < 8 * 64; j += 256)
        gs[j >> 6][j & 63] = gp[(size_t)(gbase + (j >> 6)) * 64 + (j & 63)];
    __syncthreads();
    int gg = tid >> 5, hid = tid & 31;
    float acc = bh1[hid];
    #pragma unroll
    for (int k = 0; k < 64; ++k) acc += gs[gg][k] * Wh1[k * 32 + hid];
    hs[gg][hid] = fmaxf(acc, 0.f);
    __syncthreads();
    if (tid < 80) {
        int g2 = tid / 10, oc = tid % 10;
        float o = bh2[oc];
        #pragma unroll
        for (int k = 0; k < 32; ++k) o += hs[g2][k] * Wh2[k * 10 + oc];
        out[(size_t)(gbase + g2) * 10 + oc] = o;
    }
}

extern "C" void kernel_launch(void* const* d_in, const int* in_sizes, int n_in,
                              void* d_out, int out_size, void* d_ws, size_t ws_size,
                              hipStream_t stream) {
    const int*   x    = (const int*)  d_in[0];
    const int*   ei   = (const int*)  d_in[1];
    const int*   batch= (const int*)  d_in[2];
    const float* emb  = (const float*)d_in[3];
    const float* W1   = (const float*)d_in[4];
    const float* b1   = (const float*)d_in[5];
    const float* g1   = (const float*)d_in[6];
    const float* bt1  = (const float*)d_in[7];
    const float* W2   = (const float*)d_in[8];
    const float* b2   = (const float*)d_in[9];
    const float* gbn  = (const float*)d_in[10];
    const float* bbn  = (const float*)d_in[11];
    const float* Wh1  = (const float*)d_in[12];
    const float* bh1  = (const float*)d_in[13];
    const float* Wh2  = (const float*)d_in[14];
    const float* bh2  = (const float*)d_in[15];
    float* out = (float*)d_out;

    // workspace layout
    float* wsf = (float*)d_ws;
    float*    h    = wsf;                               // N*64
    float*    agg  = h + (size_t)NNODES * 64;           // N*64
    float*    y    = agg + (size_t)NNODES * 64;         // N*128
    float*    gp   = y + (size_t)NNODES * 128;          // 512*64
    unsigned* w1p  = (unsigned*)(gp + NGRAPH * 64);     // 4*128*32
    unsigned* w2p  = w1p + 4 * 128 * 32;                // 4*64*64
    float*    misc = (float*)(w2p + 4 * 64 * 64);
    float* ssum1 = misc;        float* ssq1 = misc + 128;
    float* scale1= misc + 256;  float* shift1= misc + 384;
    float* ssum2 = misc + 512;  float* ssq2 = misc + 576;
    float* scale2= misc + 640;  float* shift2= misc + 704;

    const float invN = 1.0f / (float)NNODES;
    const int nodeT = NNODES * 16;                      // float4 elementwise threads
    const int nodeB = (nodeT + 255) / 256;
    const int gemmB = (NNODES + 127) / 128;

    // pack weights to bf16
    k_prep<<<(4 * 128 * 32 + 4 * 64 * 64 + 255) / 256, 256, 0, stream>>>(W1, W2, w1p, w2p);
    // h = emb[x]
    k_embed<<<nodeB, 256, 0, stream>>>(x, emb, h);

    for (int i = 0; i < 4; ++i) {
        k_fill0<<<(NNODES * 64 + 255) / 256, 256, 0, stream>>>(agg, NNODES * 64);
        k_fill0<<<3, 256, 0, stream>>>(misc, 768);
        k_edge<<<(NEDGES * 16) / 256, 256, 0, stream>>>(ei, h, agg);
        k_gemm1<<<gemmB, 256, 0, stream>>>(h, agg, w1p + i * 4096, b1 + i * 128,
                                           y, ssum1, ssq1);
        k_bnfold<<<1, 128, 0, stream>>>(ssum1, ssq1, g1 + i * 128, bt1 + i * 128,
                                        scale1, shift1, 128, invN);
        k_gemm2<<<gemmB, 256, 0, stream>>>(y, scale1, shift1, w2p + i * 4096, b2 + i * 64,
                                           h, ssum2, ssq2, (i < 3) ? 1 : 0);
        if (i < 3) {
            k_bnfold<<<1, 64, 0, stream>>>(ssum2, ssq2, gbn + i * 64, bbn + i * 64,
                                           scale2, shift2, 64, invN);
            k_bnapply<<<nodeB, 256, 0, stream>>>(h, scale2, shift2);
        }
    }

    k_fill0<<<(NGRAPH * 64 + 255) / 256, 256, 0, stream>>>(gp, NGRAPH * 64);
    k_pool<<<nodeB, 256, 0, stream>>>(h, batch, gp);
    k_head<<<NGRAPH / 8, 256, 0, stream>>>(gp, Wh1, bh1, Wh2, bh2, out);
}
